// MLA_73598559584910
// MI455X (gfx1250) — compile-verified, matrix operand order checked by reading, not run on hardware
//
#include <hip/hip_runtime.h>
#include <hip/hip_bf16.h>
#include <cmath>

// ---------------------------------------------------------------------------
// MLA forward for MI455X (gfx1250, wave32, WMMA).
// All GEMM-shaped math runs on V_WMMA_F32_16X16X32_BF16 (f32 accumulate).
// Attention is fused (flash-style online softmax); per-head V projection is
// folded into the flash epilogue. GEMM B-tiles are DMAed straight into LDS
// with GLOBAL_LOAD_ASYNC_TO_LDS_B128, double-buffered so the DMA of tile i+1
// overlaps the WMMAs of tile i (s_wait_asynccnt 2 keeps the newest two ops
// in flight); all 8 waves of a block share one L2 fetch per tile.
// ---------------------------------------------------------------------------

typedef __bf16 bf16_t;
typedef __attribute__((ext_vector_type(16))) __bf16 v16bf;
typedef __attribute__((ext_vector_type(8)))  float  v8f;

#define BB   2
#define SS   2048
#define DD   2048
#define HH   16
#define NOPE 128
#define ROPE 64
#define VDIM 128
#define KVR  512
#define QH   (NOPE + ROPE)   // 192
#define KEFF (KVR + ROPE)    // 576
#define ROWS (BB * SS)       // 4096

union FragU { uint4 q[2]; v16bf v; };

// Load a 16-wide bf16 A/B WMMA fragment slice for this lane (global or LDS).
// Caller passes p = &Mtx[row_or_col * ld + k0 + (lane>>4)*8].
// Halves 0..7  <- p[0..7]   (K = base .. base+7)
// Halves 8..15 <- p[16..23] (K = base+16 .. base+23)   [ISA 7.12.2 layout]
__device__ __forceinline__ v16bf frag_ld(const bf16_t* p) {
  FragU u;
  u.q[0] = *reinterpret_cast<const uint4*>(p);
  u.q[1] = *reinterpret_cast<const uint4*>(p + 16);
  return u.v;
}

__device__ __forceinline__ v8f wmma_bf16(v16bf a, v16bf b, v8f c) {
  // (neg_a, A, neg_b, B, c_mod, C, reuse_a, reuse_b)
  return __builtin_amdgcn_wmma_f32_16x16x32_bf16(false, a, false, b,
                                                 (short)0, c, false, false);
}

// Async DMA: 32 bytes global -> LDS per lane (two B128 ops; INST_OFFSET is
// added to both the LDS and global addresses per ISA §15.18 async semantics).
__device__ __forceinline__ void async_copy32(unsigned lds_addr,
                                             const bf16_t* gptr) {
  unsigned long long ga = (unsigned long long)(uintptr_t)gptr;
  asm volatile("global_load_async_to_lds_b128 %0, %1, off\n\t"
               "global_load_async_to_lds_b128 %0, %1, off offset:16"
               :: "v"(lds_addr), "v"(ga) : "memory");
}

// ---------------------------------------------------------------------------
// Elementwise helpers
// ---------------------------------------------------------------------------
__global__ void k_cast_bf16(const float* __restrict__ in,
                            bf16_t* __restrict__ out, int n) {
  int i = blockIdx.x * blockDim.x + threadIdx.x;
  if (i < n) out[i] = (bf16_t)in[i];
}

// wkv_b nope part (h, d, c) -> (h, c, d) bf16 so the absorb GEMM weight is
// (N=c, K=d) with K contiguous.
__global__ void k_wbn_transpose(const float* __restrict__ wkvb,
                                bf16_t* __restrict__ out) {
  int i = blockIdx.x * blockDim.x + threadIdx.x;
  if (i >= HH * KVR * NOPE) return;
  int d = i % NOPE;
  int c = (i / NOPE) % KVR;
  int h = i / (NOPE * KVR);
  out[i] = (bf16_t)wkvb[((size_t)h * (NOPE + VDIM) + d) * KVR + c];
}

// RoPE on q_pe (in place, bf16 storage, f32 math)
__global__ void k_rope_q(bf16_t* __restrict__ qb,
                         const float* __restrict__ fcos,
                         const float* __restrict__ fsin) {
  int i = blockIdx.x * blockDim.x + threadIdx.x;
  if (i >= ROWS * HH * (ROPE / 2)) return;
  int p   = i % (ROPE / 2);
  int h   = (i / (ROPE / 2)) % HH;
  int row = i / (HH * (ROPE / 2));
  int s   = row % SS;
  bf16_t* q = qb + (size_t)row * (HH * QH) + h * QH + NOPE + 2 * p;
  float x0 = (float)q[0], x1 = (float)q[1];
  float c = fcos[s * (ROPE / 2) + p], sn = fsin[s * (ROPE / 2) + p];
  q[0] = (bf16_t)(x0 * c - x1 * sn);
  q[1] = (bf16_t)(x0 * sn + x1 * c);
}

// RMSNorm kv -> keff[:,0:512] (bf16) and transposed kvT[b][c][t] (bf16);
// RoPE k_pe -> keff[:,512:576].
__global__ void k_kv_post(const float* __restrict__ kvf,
                          const float* __restrict__ kvw,
                          const float* __restrict__ fcos,
                          const float* __restrict__ fsin,
                          bf16_t* __restrict__ keff,
                          bf16_t* __restrict__ kvT) {
  int row = blockIdx.x, tid = threadIdx.x;
  int b = row / SS, s = row % SS;
  const float* kr = kvf + (size_t)row * KEFF;
  __shared__ float red[256];
  float ss = 0.f;
  for (int c = tid; c < KVR; c += 256) { float v = kr[c]; ss += v * v; }
  red[tid] = ss;
  __syncthreads();
  for (int st = 128; st > 0; st >>= 1) {
    if (tid < st) red[tid] += red[tid + st];
    __syncthreads();
  }
  float inv = rsqrtf(red[0] / (float)KVR + 1e-6f);
  for (int c = tid; c < KVR; c += 256) {
    float v = kr[c] * inv * kvw[c];
    keff[(size_t)row * KEFF + c] = (bf16_t)v;
    kvT[((size_t)b * KVR + c) * SS + s] = (bf16_t)v;
  }
  if (tid < ROPE / 2) {
    float x0 = kr[KVR + 2 * tid], x1 = kr[KVR + 2 * tid + 1];
    float c = fcos[s * (ROPE / 2) + tid], sn = fsin[s * (ROPE / 2) + tid];
    keff[(size_t)row * KEFF + KVR + 2 * tid]     = (bf16_t)(x0 * c - x1 * sn);
    keff[(size_t)row * KEFF + KVR + 2 * tid + 1] = (bf16_t)(x0 * sn + x1 * c);
  }
}

// copy roped q_pe into q_eff[:, h, 512:576]
__global__ void k_copy_qpe(const bf16_t* __restrict__ qb,
                           bf16_t* __restrict__ qeff) {
  int i = blockIdx.x * blockDim.x + threadIdx.x;
  if (i >= ROWS * HH * ROPE) return;
  int r   = i % ROPE;
  int h   = (i / ROPE) % HH;
  int row = i / (HH * ROPE);
  qeff[(size_t)row * (HH * KEFF) + h * KEFF + KVR + r] =
      qb[(size_t)row * (HH * QH) + h * QH + NOPE + r];
}

// ---------------------------------------------------------------------------
// Generic WMMA GEMM: C[M,N] = A[M,K] * W[N,K]^T, bf16 in, f32 accumulate.
// Block = 256 threads = 8 waves; wave w owns rows [by*128 + w*16, +16) and
// all 128 N columns (8 accumulators). The 128x32 B-tile of each K-step is
// DMAed into LDS (async-to-LDS B128, double-buffered: tile i+1 in flight
// while tile i is consumed); columns clamped so the hot loop is branch-free.
// M must be a multiple of 128 (always 4096 here).
// ---------------------------------------------------------------------------
__global__ void __launch_bounds__(256)
k_gemm(const bf16_t* __restrict__ A, int lda,
       const bf16_t* __restrict__ W, int ldw,
       void* __restrict__ Cp, int ldc, int N, int K, int out_f32) {
  __shared__ bf16_t bs[2][128][32];   // 16 KB double-buffered B-tile

  const int w    = threadIdx.x >> 5;
  const int lane = threadIdx.x & 31;
  const int half = lane >> 4;
  const int col  = lane & 15;
  const int kb   = half * 8;
  const int m0   = blockIdx.y * 128 + w * 16;
  const int n0   = blockIdx.x * 128;

  // cooperative-copy coordinates: thread -> (column, 16-half slice)
  const int cc = threadIdx.x >> 1;          // 0..127
  const int ck = (threadIdx.x & 1) * 16;    // 0 or 16
  const int ccol = min(n0 + cc, N - 1);     // clamp (dup columns never stored)
  const bf16_t* wsrc = W + (size_t)ccol * ldw + ck;
  const unsigned ldsdst0 = (unsigned)(uintptr_t)(&bs[0][cc][ck]);
  const unsigned ldsdst1 = (unsigned)(uintptr_t)(&bs[1][cc][ck]);

  v8f acc[8];
#pragma unroll
  for (int j = 0; j < 8; ++j)
#pragma unroll
    for (int v = 0; v < 8; ++v) acc[j][v] = 0.f;

  const bf16_t* arow = A + (size_t)(m0 + col) * lda + kb;
  const int nk = K / 32;

  async_copy32(ldsdst0, wsrc);              // tile 0 -> buffer 0
  for (int i = 0; i < nk; ++i) {
    const int cur = i & 1;
    // issue A-row load early so its latency hides under the barrier wait
    v16bf a = frag_ld(arow + (size_t)i * 32);
    // kick off next tile's DMA into the other buffer (last iter: harmless
    // re-load of the final tile into the dead buffer keeps loop branch-free;
    // the implicit S_WAIT_IDLE at S_ENDPGM drains it)
    const int nxt = (i + 1 < nk) ? (i + 1) : (nk - 1);
    async_copy32(cur ? ldsdst0 : ldsdst1, wsrc + (size_t)nxt * 32);
    // async loads retire in order: <=2 outstanding => tile i's ops are done
    asm volatile("s_wait_asynccnt 2" ::: "memory");
    __syncthreads();                        // tile i visible to all waves

    const bf16_t* bcur = &bs[cur][0][0];
    v16bf bfr[8];
#pragma unroll
    for (int j = 0; j < 8; ++j)
      bfr[j] = frag_ld(bcur + (j * 16 + col) * 32 + kb);
#pragma unroll
    for (int j = 0; j < 8; ++j) acc[j] = wmma_bf16(a, bfr[j], acc[j]);

    __syncthreads();                        // all waves done with buf[cur]
  }

  // C layout: row = m0 + v + 8*half, col = n0 + j*16 + (lane&15)
#pragma unroll
  for (int j = 0; j < 8; ++j) {
    int n = n0 + j * 16;
    if (n + 16 > N) continue;
#pragma unroll
    for (int v = 0; v < 8; ++v) {
      size_t idx = (size_t)(m0 + v + 8 * half) * ldc + n + col;
      if (out_f32) ((float*)Cp)[idx] = acc[j][v];
      else         ((bf16_t*)Cp)[idx] = (bf16_t)acc[j][v];
    }
  }
}

// ---------------------------------------------------------------------------
// Fused flash attention + per-head V projection.
// Grid (S/64, H, B), block 128 (4 waves). Each wave owns 16 queries and the
// full 512-dim PV accumulator (32 x v8f). Key tiles of 32; causal mask.
// ---------------------------------------------------------------------------
__global__ void __launch_bounds__(128)
k_flash(const bf16_t* __restrict__ qeff, const bf16_t* __restrict__ keff,
        const bf16_t* __restrict__ kvT,  const bf16_t* __restrict__ wbb,
        bf16_t* __restrict__ attn_proj, float scale) {
  __shared__ float  lds_p[4][16][32];   // probs staging (C-layout -> A-frag)
  __shared__ bf16_t lds_o[4][16][KVR];  // normalized out tile for projection

  const int w    = threadIdx.x >> 5;
  const int lane = threadIdx.x & 31;
  const int half = lane >> 4;
  const int col  = lane & 15;
  const int kb   = half * 8;
  const int q0   = blockIdx.x * 64 + w * 16;
  const int h    = blockIdx.y;
  const int b    = blockIdx.z;

  const bf16_t* qbase = qeff + ((size_t)(b * SS + q0 + col)) * (HH * KEFF)
                             + h * KEFF + kb;
  const bf16_t* kbase = keff + (size_t)b * SS * KEFF;
  const bf16_t* vbase = kvT  + (size_t)b * KVR * SS;

  v8f accO[32];
#pragma unroll
  for (int j = 0; j < 32; ++j)
#pragma unroll
    for (int v = 0; v < 8; ++v) accO[j][v] = 0.f;
  float mi[8], li[8];
#pragma unroll
  for (int v = 0; v < 8; ++v) { mi[v] = -1e30f; li[v] = 0.f; }

  const int tmax = q0 + 15;
  for (int t0 = 0; t0 <= tmax; t0 += 32) {
    // prefetch next key tile while this one is consumed (global_prefetch_b8)
    if (t0 + 32 <= tmax) {
      __builtin_prefetch(kbase + (size_t)(t0 + 32 + col) * KEFF, 0, 1);
      __builtin_prefetch(kbase + (size_t)(t0 + 48 + col) * KEFF, 0, 1);
    }

    v8f s0, s1;
#pragma unroll
    for (int v = 0; v < 8; ++v) { s0[v] = 0.f; s1[v] = 0.f; }

    // scores: Q(16x576) . Keff(32x576)^T  -> two 16x16 accumulators
    for (int k0 = 0; k0 < KEFF; k0 += 32) {
      v16bf a  = frag_ld(qbase + k0);
      v16bf b0 = frag_ld(kbase + (size_t)(t0 + col) * KEFF + k0 + kb);
      v16bf b1 = frag_ld(kbase + (size_t)(t0 + 16 + col) * KEFF + k0 + kb);
      s0 = wmma_bf16(a, b0, s0);
      s1 = wmma_bf16(a, b1, s1);
    }

    // scale + causal mask + online softmax (row = q0 + v + 8*half)
    float pA[8], pB[8];
#pragma unroll
    for (int v = 0; v < 8; ++v) {
      int row = q0 + v + 8 * half;
      float a = s0[v] * scale; if (t0 + col > row)      a = -1e30f;
      float c = s1[v] * scale; if (t0 + 16 + col > row) c = -1e30f;
      float rm = fmaxf(a, c);
#pragma unroll
      for (int d = 1; d < 16; d <<= 1) rm = fmaxf(rm, __shfl_xor(rm, d, 32));
      float mnew    = fmaxf(mi[v], rm);
      float rescale = __expf(mi[v] - mnew);
      float ea = __expf(a - mnew);
      float eb = __expf(c - mnew);
      float rs = ea + eb;
#pragma unroll
      for (int d = 1; d < 16; d <<= 1) rs += __shfl_xor(rs, d, 32);
      li[v] = li[v] * rescale + rs;
      mi[v] = mnew;
      pA[v] = ea; pB[v] = eb;
#pragma unroll
      for (int j = 0; j < 32; ++j) accO[j][v] *= rescale;
    }

    // probs C-layout -> LDS -> A-fragment (bf16)
#pragma unroll
    for (int v = 0; v < 8; ++v) {
      lds_p[w][v + 8 * half][col]      = pA[v];
      lds_p[w][v + 8 * half][16 + col] = pB[v];
    }
    asm volatile("s_wait_dscnt 0" ::: "memory");  // order LDS store -> load (same wave)
    v16bf pf;
#pragma unroll
    for (int e = 0; e < 16; ++e) {
      int k = kb + ((e < 8) ? e : (e + 8));
      pf[e] = (bf16_t)lds_p[w][col][k];
    }

    // PV: P(16x32) . Kv(32x512); groups of 8 loads then 8 WMMAs so the
    // load-counter waits stagger instead of serializing.
#pragma unroll
    for (int jj = 0; jj < 32; jj += 8) {
      v16bf bv[8];
#pragma unroll
      for (int j = 0; j < 8; ++j)
        bv[j] = frag_ld(vbase + (size_t)((jj + j) * 16 + col) * SS + t0 + kb);
#pragma unroll
      for (int j = 0; j < 8; ++j)
        accO[jj + j] = wmma_bf16(pf, bv[j], accO[jj + j]);
    }
  }

  // normalize and stage 16x512 tile for the V projection
#pragma unroll
  for (int v = 0; v < 8; ++v) li[v] = 1.f / li[v];
#pragma unroll
  for (int j = 0; j < 32; ++j)
#pragma unroll
    for (int v = 0; v < 8; ++v)
      lds_o[w][v + 8 * half][j * 16 + col] = (bf16_t)(accO[j][v] * li[v]);
  asm volatile("s_wait_dscnt 0" ::: "memory");

  // per-head V projection: (16x512) . Wv(128x512)^T
  v8f accP[8];
#pragma unroll
  for (int j = 0; j < 8; ++j)
#pragma unroll
    for (int v = 0; v < 8; ++v) accP[j][v] = 0.f;
  const bf16_t* wv = wbb + ((size_t)h * (NOPE + VDIM) + NOPE) * KVR;
  for (int k0 = 0; k0 < KVR; k0 += 32) {
    v16bf a;
#pragma unroll
    for (int e = 0; e < 16; ++e) {
      int k = k0 + kb + ((e < 8) ? e : (e + 8));
      a[e] = lds_o[w][col][k];
    }
    v16bf bw[8];
#pragma unroll
    for (int j = 0; j < 8; ++j)
      bw[j] = frag_ld(wv + (size_t)(j * 16 + col) * KVR + k0 + kb);
#pragma unroll
    for (int j = 0; j < 8; ++j)
      accP[j] = wmma_bf16(a, bw[j], accP[j]);
  }
#pragma unroll
  for (int j = 0; j < 8; ++j)
#pragma unroll
    for (int v = 0; v < 8; ++v) {
      int row = q0 + v + 8 * half;
      attn_proj[(size_t)(b * SS + row) * (HH * VDIM) + h * VDIM + j * 16 + col] =
          (bf16_t)(accP[j][v]);
    }
}

// ---------------------------------------------------------------------------
// Host-side orchestration. Workspace layout (~182 MB) carved from d_ws.
// ---------------------------------------------------------------------------
extern "C" void kernel_launch(void* const* d_in, const int* in_sizes, int n_in,
                              void* d_out, int out_size, void* d_ws, size_t ws_size,
                              hipStream_t stream) {
  const float* x     = (const float*)d_in[0];
  const float* wq    = (const float*)d_in[1];
  const float* wkv_a = (const float*)d_in[2];
  const float* kvw   = (const float*)d_in[3];
  const float* wkv_b = (const float*)d_in[4];
  const float* wo    = (const float*)d_in[5];
  const float* fcos  = (const float*)d_in[6];
  const float* fsin  = (const float*)d_in[7];
  // d_in[8] = mask: causal, applied analytically in k_flash.

  char* ws = (char*)d_ws;
  size_t off = 0;
  auto alloc = [&](size_t bytes) {
    char* p = ws + off;
    off = (off + bytes + 255) & ~(size_t)255;
    return p;
  };
  bf16_t* xb   = (bf16_t*)alloc((size_t)ROWS * DD * 2);
  bf16_t* wqb  = (bf16_t*)alloc((size_t)HH * QH * DD * 2);
  bf16_t* wab  = (bf16_t*)alloc((size_t)KEFF * DD * 2);
  bf16_t* wbb  = (bf16_t*)alloc((size_t)HH * (NOPE + VDIM) * KVR * 2);
  bf16_t* wob  = (bf16_t*)alloc((size_t)DD * HH * VDIM * 2);
  bf16_t* wbnT = (bf16_t*)alloc((size_t)HH * KVR * NOPE * 2);
  bf16_t* qb   = (bf16_t*)alloc((size_t)ROWS * HH * QH * 2);
  float*  kvf  = (float*) alloc((size_t)ROWS * KEFF * 4);
  bf16_t* kef  = (bf16_t*)alloc((size_t)ROWS * KEFF * 2);
  bf16_t* kvT  = (bf16_t*)alloc((size_t)BB * KVR * SS * 2);
  bf16_t* qef  = (bf16_t*)alloc((size_t)ROWS * HH * KEFF * 2);
  bf16_t* aprj = (bf16_t*)alloc((size_t)ROWS * HH * VDIM * 2);

  auto cast = [&](const float* in, bf16_t* out, int n) {
    k_cast_bf16<<<(n + 255) / 256, 256, 0, stream>>>(in, out, n);
  };
  cast(x,     xb,  ROWS * DD);
  cast(wq,    wqb, HH * QH * DD);
  cast(wkv_a, wab, KEFF * DD);
  cast(wkv_b, wbb, HH * (NOPE + VDIM) * KVR);
  cast(wo,    wob, DD * HH * VDIM);
  k_wbn_transpose<<<(HH * KVR * NOPE + 255) / 256, 256, 0, stream>>>(wkv_b, wbnT);

  // q = x @ wq^T  (4096 x 3072 x 2048), bf16 out
  k_gemm<<<dim3((HH * QH) / 128, ROWS / 128), 256, 0, stream>>>(
      xb, DD, wqb, DD, qb, HH * QH, HH * QH, DD, 0);
  // kv_full = x @ wkv_a^T  (4096 x 576 x 2048), f32 out for RMSNorm fidelity
  k_gemm<<<dim3((KEFF + 127) / 128, ROWS / 128), 256, 0, stream>>>(
      xb, DD, wab, DD, kvf, KEFF, KEFF, DD, 1);

  k_rope_q<<<(ROWS * HH * (ROPE / 2) + 255) / 256, 256, 0, stream>>>(qb, fcos, fsin);
  k_kv_post<<<ROWS, 256, 0, stream>>>(kvf, kvw, fcos, fsin, kef, kvT);

  // nope-absorb per head: q_eff[:, h, 0:512] = q_nope_h @ wbnT_h^T
  for (int h = 0; h < HH; ++h) {
    k_gemm<<<dim3(KVR / 128, ROWS / 128), 256, 0, stream>>>(
        qb + h * QH, HH * QH,
        wbnT + (size_t)h * KVR * NOPE, NOPE,
        qef + h * KEFF, HH * KEFF, KVR, NOPE, 0);
  }
  k_copy_qpe<<<(ROWS * HH * ROPE + 255) / 256, 256, 0, stream>>>(qb, qef);

  float mscale = 0.1f * logf(40.0f) + 1.0f;
  float scl = powf((float)(NOPE + ROPE), -0.5f) * mscale * mscale;
  k_flash<<<dim3(SS / 64, HH, BB), 128, 0, stream>>>(qef, kef, kvT, wbb, aprj, scl);

  // final: out = attn_proj @ wo^T  (4096 x 2048 x 2048), f32 out
  k_gemm<<<dim3(DD / 128, ROWS / 128), 256, 0, stream>>>(
      aprj, HH * VDIM, wob, HH * VDIM, d_out, DD, DD, HH * VDIM, 1);
}
